// model_encdec_61443802137200
// MI455X (gfx1250) — compile-verified, matrix operand order checked by reading, not run on hardware
//
#include <hip/hip_runtime.h>
#include <hip/hip_bf16.h>

// ---------------------------------------------------------------------------
// Shapes (from reference):
//   B=2048, N_MEM=131072, PAST_LEN=8, D=64, K_TOP=200, S=64, DK=128, HID=256
// Output layout (flat, return order):
//   state_past            [2048 x 128]
//   state_past_selector   [2048 x 128]
//   mem_sel               [2048 x 200 x 128]
//   memory_past_selector  [2048 x 200 x 128]
// ---------------------------------------------------------------------------

#define B_ROWS 2048
#define NMEM   131072
#define DKDIM  128
#define HIDDIM 256
#define KTOP   200
#define SEGS   64
#define NB     512   // histogram buckets over [-1,1]
#define CAP    512   // candidate capacity per row (power of 2 for bitonic)

typedef _Float16 v16h __attribute__((ext_vector_type(16)));
typedef _Float16 v8h  __attribute__((ext_vector_type(8)));
typedef _Float16 v4h  __attribute__((ext_vector_type(4)));
typedef float    v8f  __attribute__((ext_vector_type(8)));

// ---- WMMA fragment builders (CDNA5 16-bit A/B layout, wave32) --------------
// A (16x32 f16): lane l -> row m=l&15; halves e=0..7 -> K = c*32 + (l>>4)*8 + e
//                              e=8..15 -> K = c*32 + 16 + (l>>4)*8 + (e-8)
// B (32x16 f16): lane l -> col n=l&15 with identical K packing (row n of a
//                NxK-transposed operand is contiguous -> two 16B loads).
__device__ __forceinline__ v16h frag_h16(const _Float16* base, int ld, int c, int lane) {
    int m = lane & 15, hi = lane >> 4;
    const _Float16* p = base + m * ld + c * 32 + hi * 8;
    v8h a = *(const v8h*)p;
    v8h b = *(const v8h*)(p + 16);
    v16h r;
#pragma unroll
    for (int i = 0; i < 8; ++i) { r[i] = a[i]; r[i + 8] = b[i]; }
    return r;
}

__device__ __forceinline__ v16h frag_f32cvt(const float* base, int ld, int c, int lane) {
    int m = lane & 15, hi = lane >> 4;
    const float* p = base + m * ld + c * 32 + hi * 8;
    v16h r;
#pragma unroll
    for (int i = 0; i < 8; ++i) { r[i] = (_Float16)p[i]; r[i + 8] = (_Float16)p[16 + i]; }
    return r;
}

// ---------------------------------------------------------------------------
__global__ void zero_f32_kernel(float* p, int n) {
    int i = blockIdx.x * 256 + threadIdx.x;
    if (i < n) p[i] = 0.f;
}

// past/abs_past encoders (2048x16 @ 16x64) + segment accumulation
__global__ void encode_kernel(const float* __restrict__ past, const float* __restrict__ absp,
                              const float* __restrict__ w_np, const float* __restrict__ b_np,
                              const float* __restrict__ w_ap, const float* __restrict__ b_ap,
                              const int* __restrict__ sse,
                              float* __restrict__ state_past, float* __restrict__ abs_state,
                              float* __restrict__ segsum, float* __restrict__ segcnt) {
    int idx = blockIdx.x * 256 + threadIdx.x;
    if (idx >= B_ROWS * 64) return;
    int b = idx >> 6, d = idx & 63;
    float acc = b_np[d];
#pragma unroll
    for (int k = 0; k < 16; ++k) acc = fmaf(past[b * 16 + k], w_np[k * 64 + d], acc);
    state_past[b * 128 + d] = fmaxf(acc, 0.f);
    float a2 = b_ap[d];
#pragma unroll
    for (int k = 0; k < 16; ++k) a2 = fmaf(absp[b * 16 + k], w_ap[k * 64 + d], a2);
    a2 = fmaxf(a2, 0.f);
    abs_state[b * 64 + d] = a2;
    // seg = largest s with starts[s] <= b  (searchsorted right - 1)
    int lo = 0, hi = SEGS - 1;
    while (lo < hi) { int mid = (lo + hi + 1) >> 1; if (sse[2 * mid] <= b) lo = mid; else hi = mid - 1; }
    atomicAdd(&segsum[lo * 64 + d], a2);
    if (d == 0) atomicAdd(&segcnt[lo], 1.0f);
}

// social = relu([abs_state, pooled, end_pose] @ w_sp + b_sp) -> state_past cols 64..127
__global__ void social_kernel(const float* __restrict__ abs_state, const float* __restrict__ segsum,
                              const float* __restrict__ segcnt, const float* __restrict__ end_pose,
                              const float* __restrict__ w_sp, const float* __restrict__ b_sp,
                              const int* __restrict__ sse, float* __restrict__ state_past) {
    int idx = blockIdx.x * 256 + threadIdx.x;
    if (idx >= B_ROWS * 64) return;
    int b = idx >> 6, d = idx & 63;
    int lo = 0, hi = SEGS - 1;
    while (lo < hi) { int mid = (lo + hi + 1) >> 1; if (sse[2 * mid] <= b) lo = mid; else hi = mid - 1; }
    float inv = 1.f / fmaxf(segcnt[lo], 1.f);
    float acc = b_sp[d];
#pragma unroll 8
    for (int k = 0; k < 64; ++k) acc = fmaf(abs_state[b * 64 + k], w_sp[k * 64 + d], acc);
#pragma unroll 8
    for (int k = 0; k < 64; ++k) acc = fmaf(segsum[lo * 64 + k] * inv, w_sp[(64 + k) * 64 + d], acc);
    acc = fmaf(end_pose[b * 2 + 0], w_sp[128 * 64 + d], acc);
    acc = fmaf(end_pose[b * 2 + 1], w_sp[129 * 64 + d], acc);
    state_past[b * 128 + 64 + d] = fmaxf(acc, 0.f);
}

// per-row (width 128) L2 normalize fp32 -> f16, one wave per row
__global__ void l2norm_f16_kernel(const float* __restrict__ in, _Float16* __restrict__ out, int rows) {
    int gw = (int)((blockIdx.x * (unsigned)blockDim.x + threadIdx.x) >> 5);
    int lane = threadIdx.x & 31;
    if (gw >= rows) return;
    float4 v = ((const float4*)(in + (long)gw * DKDIM))[lane];
    float ss = v.x * v.x + v.y * v.y + v.z * v.z + v.w * v.w;
#pragma unroll
    for (int o = 16; o > 0; o >>= 1) ss += __shfl_xor(ss, o, 32);
    float inv = rsqrtf(fmaxf(ss, 1e-24f));
    v4h h; h[0] = (_Float16)(v.x * inv); h[1] = (_Float16)(v.y * inv);
    h[2] = (_Float16)(v.z * inv); h[3] = (_Float16)(v.w * inv);
    *(v4h*)(out + (long)gw * DKDIM + lane * 4) = h;
}

// W (KxN fp32, row-major) -> Wt (NxK f16) so B-columns are contiguous
__global__ void wt_convert_kernel(const float* __restrict__ w, int K, int N, _Float16* __restrict__ out) {
    int i = blockIdx.x * 256 + threadIdx.x;
    if (i >= K * N) return;
    int n = i / K, k = i - n * K;
    out[n * K + k] = (_Float16)w[k * N + n];
}

// ---------------------------------------------------------------------------
// Fused similarity + top-K + gather.
// Block: 16 query rows, 256 threads (8 waves). Each wave scans a stripe of the
// 131072 memory rows in 16-row WMMA tiles. Pass A builds a per-row score
// histogram; pass B collects candidates >= threshold bucket; bitonic sort
// descending; emit top-200 and gather fp32 memory rows to mem_sel.
// ---------------------------------------------------------------------------
__global__ void __launch_bounds__(256) sim_topk_kernel(const _Float16* __restrict__ stateN,
                                                       const _Float16* __restrict__ memN,
                                                       const float* __restrict__ memPast,
                                                       float* __restrict__ msel) {
    __shared__ __align__(16) _Float16 q[16][136];
    __shared__ unsigned hist[16][NB];
    __shared__ float cs[16][CAP];
    __shared__ int   ci[16][CAP];
    __shared__ int   cnt[16];
    __shared__ int   thr[16];

    int t = threadIdx.x, lane = t & 31, wv = t >> 5;
    int qbase = blockIdx.x * 16;

    for (int i = t; i < 16 * DKDIM; i += 256) q[i >> 7][i & 127] = stateN[(long)qbase * DKDIM + i];
    unsigned* hflat = &hist[0][0];
    for (int i = t; i < 16 * NB; i += 256) hflat[i] = 0;
    if (t < 16) cnt[t] = 0;
    __syncthreads();

    v16h a[4];
#pragma unroll
    for (int c = 0; c < 4; ++c) a[c] = frag_h16(&q[0][0], 136, c, lane);

    const int NT = NMEM / 16;
    // ---- pass A: histogram ----
    for (int tile = wv; tile < NT; tile += 8) {
        const _Float16* mb = memN + (long)tile * 16 * DKDIM;
        __builtin_prefetch(mb + 8 * 16 * DKDIM, 0, 0);
        v8f acc = {};
#pragma unroll
        for (int c = 0; c < 4; ++c)
            acc = __builtin_amdgcn_wmma_f32_16x16x32_f16(false, a[c], false,
                    frag_h16(mb, DKDIM, c, lane), (short)0, acc, false, false);
#pragma unroll
        for (int r = 0; r < 8; ++r) {
            int row = r + (lane >> 4) * 8;
            float s = acc[r];
            int b = (int)((s + 1.0f) * (NB * 0.5f));
            b = b < 0 ? 0 : (b > NB - 1 ? NB - 1 : b);
            atomicAdd(&hist[row][b], 1u);
        }
    }
    __syncthreads();
    if (t < 16) {
        int cum = 0, b = NB - 1;
        for (; b > 0; --b) { cum += (int)hist[t][b]; if (cum >= KTOP) break; }
        thr[t] = b;
    }
    __syncthreads();

    // ---- pass B: collect candidates (memory is L2-hot; recompute is cheap) ----
    for (int tile = wv; tile < NT; tile += 8) {
        const _Float16* mb = memN + (long)tile * 16 * DKDIM;
        __builtin_prefetch(mb + 8 * 16 * DKDIM, 0, 0);
        v8f acc = {};
#pragma unroll
        for (int c = 0; c < 4; ++c)
            acc = __builtin_amdgcn_wmma_f32_16x16x32_f16(false, a[c], false,
                    frag_h16(mb, DKDIM, c, lane), (short)0, acc, false, false);
#pragma unroll
        for (int r = 0; r < 8; ++r) {
            int row = r + (lane >> 4) * 8;
            float s = acc[r];
            int b = (int)((s + 1.0f) * (NB * 0.5f));
            b = b < 0 ? 0 : (b > NB - 1 ? NB - 1 : b);
            if (b >= thr[row]) {
                int pos = atomicAdd(&cnt[row], 1);
                if (pos < CAP) { cs[row][pos] = s; ci[row][pos] = tile * 16 + (lane & 15); }
            }
        }
    }
    __syncthreads();

    // pad unused slots, each wave owns 2 rows
    for (int rr = 0; rr < 2; ++rr) {
        int row = wv * 2 + rr;
        int nC = cnt[row]; if (nC > CAP) nC = CAP;
        for (int i = lane; i < CAP; i += 32)
            if (i >= nC) { cs[row][i] = -3.0e38f; ci[row][i] = 0; }
    }
    __syncthreads();

    // bitonic sort, descending (uniform trip counts -> block syncs are safe)
    for (int k2 = 2; k2 <= CAP; k2 <<= 1) {
        for (int j = k2 >> 1; j > 0; j >>= 1) {
            for (int rr = 0; rr < 2; ++rr) {
                int row = wv * 2 + rr;
                for (int i = lane; i < CAP; i += 32) {
                    int ixj = i ^ j;
                    if (ixj > i) {
                        float x = cs[row][i], y = cs[row][ixj];
                        bool up = ((i & k2) == 0);
                        if ((x < y) == up) {
                            cs[row][i] = y; cs[row][ixj] = x;
                            int ti = ci[row][i]; ci[row][i] = ci[row][ixj]; ci[row][ixj] = ti;
                        }
                    }
                }
            }
            __syncthreads();
        }
    }

    // gather: mem_sel[qbase+row][j] = memory_past[ci[row][j]]  (fp32, float4)
    const float4* m4 = (const float4*)memPast;
    float4* o4 = (float4*)msel;
    for (int i = t; i < 16 * KTOP * 32; i += 256) {
        int ch  = i & 31;
        int j   = (i >> 5) % KTOP;
        int row = (i >> 5) / KTOP;
        int mrow = ci[row][j];
        o4[((long)(qbase + row) * KTOP + j) * 32 + ch] = m4[(long)mrow * 32 + ch];
    }
}

// ---------------------------------------------------------------------------
// 3-layer MLP: relu(relu(X@W1+b1)@W2+b2)@W3+b3, 128->256->256->128.
// 128 rows per block (16 per wave). Weights pre-transposed to NxK f16.
// Hidden activations staged per-wave in LDS (stride 264 to dodge DS bank
// conflicts on the fragment b128 reads).
// ---------------------------------------------------------------------------
#define LDH 264
__global__ void __launch_bounds__(256) mlp_kernel(const float* __restrict__ X,
                                                  const _Float16* __restrict__ w1t, const float* __restrict__ b1,
                                                  const _Float16* __restrict__ w2t, const float* __restrict__ b2,
                                                  const _Float16* __restrict__ w3t, const float* __restrict__ b3,
                                                  float* __restrict__ Y) {
    __shared__ __align__(16) _Float16 hA[8][16][LDH];
    __shared__ __align__(16) _Float16 hB[8][16][LDH];
    int lane = threadIdx.x & 31, wv = threadIdx.x >> 5;
    long rowBase = (long)blockIdx.x * 128 + wv * 16;
    int n = lane & 15, hi = lane >> 4;

    v16h a[8];
#pragma unroll
    for (int c = 0; c < 4; ++c) a[c] = frag_f32cvt(X + rowBase * DKDIM, DKDIM, c, lane);

    // layer 1: 16x128 @ 128x256
    for (int nt = 0; nt < 16; ++nt) {
        v8f acc = {};
#pragma unroll
        for (int c = 0; c < 4; ++c)
            acc = __builtin_amdgcn_wmma_f32_16x16x32_f16(false, a[c], false,
                    frag_h16(w1t + nt * 16 * DKDIM, DKDIM, c, lane), (short)0, acc, false, false);
        float bias = b1[nt * 16 + n];
#pragma unroll
        for (int r = 0; r < 8; ++r) {
            float v = acc[r] + bias; v = v > 0.f ? v : 0.f;
            hA[wv][r + hi * 8][nt * 16 + n] = (_Float16)v;
        }
    }
    __syncthreads();

    // layer 2: 16x256 @ 256x256
#pragma unroll
    for (int c = 0; c < 8; ++c) a[c] = frag_h16(&hA[wv][0][0], LDH, c, lane);
    for (int nt = 0; nt < 16; ++nt) {
        v8f acc = {};
#pragma unroll
        for (int c = 0; c < 8; ++c)
            acc = __builtin_amdgcn_wmma_f32_16x16x32_f16(false, a[c], false,
                    frag_h16(w2t + nt * 16 * HIDDIM, HIDDIM, c, lane), (short)0, acc, false, false);
        float bias = b2[nt * 16 + n];
#pragma unroll
        for (int r = 0; r < 8; ++r) {
            float v = acc[r] + bias; v = v > 0.f ? v : 0.f;
            hB[wv][r + hi * 8][nt * 16 + n] = (_Float16)v;
        }
    }
    __syncthreads();

    // layer 3: 16x256 @ 256x128 -> fp32 out
#pragma unroll
    for (int c = 0; c < 8; ++c) a[c] = frag_h16(&hB[wv][0][0], LDH, c, lane);
    for (int nt = 0; nt < 8; ++nt) {
        v8f acc = {};
#pragma unroll
        for (int c = 0; c < 8; ++c)
            acc = __builtin_amdgcn_wmma_f32_16x16x32_f16(false, a[c], false,
                    frag_h16(w3t + nt * 16 * HIDDIM, HIDDIM, c, lane), (short)0, acc, false, false);
        float bias = b3[nt * 16 + n];
#pragma unroll
        for (int r = 0; r < 8; ++r)
            Y[(rowBase + r + hi * 8) * DKDIM + nt * 16 + n] = acc[r] + bias;
    }
}

// ---------------------------------------------------------------------------
extern "C" void kernel_launch(void* const* d_in, const int* in_sizes, int n_in,
                              void* d_out, int out_size, void* d_ws, size_t ws_size,
                              hipStream_t stream) {
    const float* past     = (const float*)d_in[0];
    const float* absp     = (const float*)d_in[1];
    const float* end_pose = (const float*)d_in[2];
    /* d_in[3] future: unused by the reference outputs */
    const int*   sse      = (const int*)d_in[4];
    const float* memory   = (const float*)d_in[5];
    const float* w_np = (const float*)d_in[6];  const float* b_np = (const float*)d_in[7];
    const float* w_ap = (const float*)d_in[8];  const float* b_ap = (const float*)d_in[9];
    const float* w_sp = (const float*)d_in[10]; const float* b_sp = (const float*)d_in[11];
    const float* qw1 = (const float*)d_in[12];  const float* qb1 = (const float*)d_in[13];
    const float* qw2 = (const float*)d_in[14];  const float* qb2 = (const float*)d_in[15];
    const float* qw3 = (const float*)d_in[16];  const float* qb3 = (const float*)d_in[17];
    const float* mw1 = (const float*)d_in[18];  const float* mb1 = (const float*)d_in[19];
    const float* mw2 = (const float*)d_in[20];  const float* mb2 = (const float*)d_in[21];
    const float* mw3 = (const float*)d_in[22];  const float* mb3 = (const float*)d_in[23];

    float* out  = (float*)d_out;
    float* sp   = out;                                    // state_past
    float* sps  = sp  + (size_t)B_ROWS * DKDIM;           // state_past_selector
    float* msel = sps + (size_t)B_ROWS * DKDIM;           // mem_sel
    float* mps  = msel + (size_t)B_ROWS * KTOP * DKDIM;   // memory_past_selector

    char* w = (char*)d_ws;
    _Float16* memN = (_Float16*)w;  w += (size_t)NMEM * DKDIM * 2;
    _Float16* stN  = (_Float16*)w;  w += (size_t)B_ROWS * DKDIM * 2;
    float* abs_state = (float*)w;   w += (size_t)B_ROWS * 64 * 4;
    float* segsum    = (float*)w;   w += SEGS * 64 * 4;
    float* segcnt    = (float*)w;   w += SEGS * 4;
    _Float16* qw1t = (_Float16*)w;  w += DKDIM * HIDDIM * 2;
    _Float16* qw2t = (_Float16*)w;  w += HIDDIM * HIDDIM * 2;
    _Float16* qw3t = (_Float16*)w;  w += HIDDIM * DKDIM * 2;
    _Float16* mw1t = (_Float16*)w;  w += DKDIM * HIDDIM * 2;
    _Float16* mw2t = (_Float16*)w;  w += HIDDIM * HIDDIM * 2;
    _Float16* mw3t = (_Float16*)w;  w += HIDDIM * DKDIM * 2;

    // phase 0: zero segment accumulators
    zero_f32_kernel<<<(SEGS * 64 + SEGS + 255) / 256, 256, 0, stream>>>(segsum, SEGS * 64 + SEGS);

    // phase 1: encoders + pooling + social
    encode_kernel<<<(B_ROWS * 64) / 256, 256, 0, stream>>>(past, absp, w_np, b_np, w_ap, b_ap,
                                                           sse, sp, abs_state, segsum, segcnt);
    social_kernel<<<(B_ROWS * 64) / 256, 256, 0, stream>>>(abs_state, segsum, segcnt, end_pose,
                                                           w_sp, b_sp, sse, sp);

    // phase 2: l2-normalize (f16) state + memory; convert+transpose MLP weights
    l2norm_f16_kernel<<<B_ROWS / 4, 128, 0, stream>>>(sp, stN, B_ROWS);
    l2norm_f16_kernel<<<NMEM / 4, 128, 0, stream>>>(memory, memN, NMEM);
    wt_convert_kernel<<<(DKDIM * HIDDIM + 255) / 256, 256, 0, stream>>>(qw1, DKDIM, HIDDIM, qw1t);
    wt_convert_kernel<<<(HIDDIM * HIDDIM + 255) / 256, 256, 0, stream>>>(qw2, HIDDIM, HIDDIM, qw2t);
    wt_convert_kernel<<<(HIDDIM * DKDIM + 255) / 256, 256, 0, stream>>>(qw3, HIDDIM, DKDIM, qw3t);
    wt_convert_kernel<<<(DKDIM * HIDDIM + 255) / 256, 256, 0, stream>>>(mw1, DKDIM, HIDDIM, mw1t);
    wt_convert_kernel<<<(HIDDIM * HIDDIM + 255) / 256, 256, 0, stream>>>(mw2, HIDDIM, HIDDIM, mw2t);
    wt_convert_kernel<<<(HIDDIM * DKDIM + 255) / 256, 256, 0, stream>>>(mw3, HIDDIM, DKDIM, mw3t);

    // phase 3: fused similarity + top-200 + gather mem_sel
    sim_topk_kernel<<<B_ROWS / 16, 256, 0, stream>>>(stN, memN, memory, msel);

    // phase 4: selector MLPs (WMMA)
    mlp_kernel<<<B_ROWS / 128, 256, 0, stream>>>(sp, qw1t, qb1, qw2t, qb2, qw3t, qb3, sps);
    mlp_kernel<<<(B_ROWS * KTOP) / 128, 256, 0, stream>>>(msel, mw1t, mb1, mw2t, mb2, mw3t, mb3, mps);
}
